// HeteroGATLayer_50414326120542
// MI455X (gfx1250) — matrix operand order repaired and checked by reading, not compiled, not run on hardware
//
#include <hip/hip_runtime.h>
#include <math.h>

// ---------------- problem constants ----------------
constexpr int NN   = 50000;      // nodes per type
constexpr int KD   = 256;        // D_IN
constexpr int HH   = 4;          // heads
constexpr int DO   = 32;         // D_OUT per head
constexpr int HD   = HH * DO;    // 128
constexpr int EE   = 800000;     // edges per relation
constexpr int NCAT = 176;        // 128 (fc) + 32 (loop) + 4 (el) + 4 (er) + 8 pad
constexpr int NT_TILES = NCAT / 16;   // 11
constexpr int KSTEPS   = KD / 32;     // 8
constexpr int PNT      = KSTEPS * NT_TILES * 32 * 8;  // 22528 u32 per ntype (B fragments)

typedef __attribute__((ext_vector_type(16))) __bf16 v16bf;
typedef __attribute__((ext_vector_type(8)))  float  v8f;

// ---------------- helpers ----------------
__device__ __forceinline__ unsigned short f2bf(float x) {
    unsigned b = __float_as_uint(x);
    unsigned r = b + 0x7FFFu + ((b >> 16) & 1u);   // round-to-nearest-even
    return (unsigned short)(r >> 16);
}
__device__ __forceinline__ unsigned pack_bf2(float lo, float hi) {
    return (unsigned)f2bf(lo) | ((unsigned)f2bf(hi) << 16);
}
__device__ __forceinline__ void atomicMaxFloat(float* addr, float val) {
    if (val >= 0.f) atomicMax((int*)addr, __float_as_int(val));
    else            atomicMin((unsigned int*)addr, (unsigned int)__float_as_int(val));
}
__device__ __forceinline__ float eluf(float x) {
    return x > 0.f ? x : __expf(x) - 1.f;
}

// ---------------- K0: build B-fragment-packed cat-weight matrices ----------------
// Logical Wcat (256 x 176): ntype 0 (user): [fc_r1 | loop | wl1 | wr2],
//                           ntype 1 (item): [fc_r2 | loop | wl2 | wr1]
// Stored directly in this file's WMMA B-fragment order (contiguous k = 16h + s):
//   Bfrag[nt][ks][tile][lane][d]  (u32 = bf16 pair, K = 2p, 2p+1)
//   with n = tile*16 + (lane&15),  p = ks*16 + (lane>=16 ? 8 : 0) + d
__device__ __forceinline__ float welem(int k, int n,
                                       const float* fcL, const float* al,
                                       const float* fcR, const float* ar,
                                       const float* lw) {
    if (n < HD)       return fcL[k * HD + n];
    if (n < HD + DO)  return lw[k * DO + (n - HD)];
    if (n < HD + DO + HH) {
        int h = n - (HD + DO); float s = 0.f;
        for (int d = 0; d < DO; ++d) s += fcL[k * HD + h * DO + d] * al[h * DO + d];
        return s;
    }
    if (n < HD + DO + 2 * HH) {
        int h = n - (HD + DO + HH); float s = 0.f;
        for (int d = 0; d < DO; ++d) s += fcR[k * HD + h * DO + d] * ar[h * DO + d];
        return s;
    }
    return 0.f;
}

__global__ __launch_bounds__(256)
void pack_w_kernel(const float* __restrict__ fc1, const float* __restrict__ fc2,
                   const float* __restrict__ al1, const float* __restrict__ ar1,
                   const float* __restrict__ al2, const float* __restrict__ ar2,
                   const float* __restrict__ lw, unsigned* __restrict__ Wp) {
    int idx = blockIdx.x * blockDim.x + threadIdx.x;
    if (idx >= 2 * PNT) return;
    int nt = idx / PNT;
    int r  = idx - nt * PNT;
    int d    = r & 7;
    int lane = (r >> 3) & 31;
    int kt   = r >> 8;            // (ks*11 + t), since 32 lanes * 8 dwords = 256
    int t    = kt % NT_TILES;
    int ks   = kt / NT_TILES;
    int n = t * 16 + (lane & 15);
    int p = ks * 16 + ((lane & 16) ? 8 : 0) + d;

    const float* fcL = nt ? fc2 : fc1;
    const float* al  = nt ? al2 : al1;
    const float* fcR = nt ? fc1 : fc2;
    const float* ar  = nt ? ar1 : ar2;
    float w0 = welem(2 * p,     n, fcL, al, fcR, ar, lw);
    float w1 = welem(2 * p + 1, n, fcL, al, fcR, ar, lw);
    Wp[idx] = pack_bf2(w0, w1);
}

// ---------------- K1: init accumulators ----------------
__global__ __launch_bounds__(256)
void init_kernel(float* __restrict__ out,
                 float* __restrict__ emax1, float* __restrict__ emax2,
                 float* __restrict__ den1,  float* __restrict__ den2) {
    int idx = blockIdx.x * blockDim.x + threadIdx.x;
    if (idx < 2 * NN * HD) out[idx] = 0.f;
    if (idx < NN * HH) {
        float ninf = __int_as_float(0xff800000);
        emax1[idx] = ninf; emax2[idx] = ninf;
        den1[idx] = 0.f;   den2[idx] = 0.f;
    }
}

// ---------------- K2: WMMA GEMM  C(NN x NCAT) = X(NN x 256) @ Wcat(256 x NCAT) ----------------
// Each wave owns a 16-row band; A staged per-wave through LDS (transpose into
// fragment layout). All 11 B fragments for a k-step are loaded into distinct,
// simultaneously-live register buffers (one big load clause), then the 11 WMMAs
// issue back-to-back — no per-tile load drain. No cross-wave LDS sharing ->
// no barriers needed.
__global__ __launch_bounds__(256)
void gemm_cat_kernel(const float* __restrict__ Xu, const float* __restrict__ Xi,
                     const unsigned* __restrict__ Wp,
                     float* __restrict__ Cu, float* __restrict__ Ci) {
    __shared__ unsigned aLds[128 * 16];      // [row][kpair]  (bf16 x2 per entry)

    const int nt = blockIdx.y;
    const float*    X = nt ? Xi : Xu;
    float*          C = nt ? Ci : Cu;
    const unsigned* W = Wp + nt * PNT;

    const int tid  = threadIdx.x;
    const int wave = tid >> 5;
    const int lane = tid & 31;
    const int mBase = blockIdx.x * 128;

    v8f acc[NT_TILES];
    #pragma unroll
    for (int t = 0; t < NT_TILES; ++t)
        #pragma unroll
        for (int j = 0; j < 8; ++j) acc[t][j] = 0.f;

    const int arow  = (wave << 4) + (lane & 15);
    const int pbase = (lane >= 16) ? 8 : 0;

    union BF { v16bf v; uint4 q[2]; };

    for (int ks = 0; ks < KSTEPS; ++ks) {
        // ---- issue ALL 11 B fragments first: one clause, distinct live buffers ----
        BF bufs[NT_TILES];
        {
            const uint4* bbase =
                (const uint4*)(W + (((ks * NT_TILES) << 5) + lane) * 8);
            #pragma unroll
            for (int t = 0; t < NT_TILES; ++t) {
                bufs[t].q[0] = bbase[t * 64];        // tile stride = 32 lanes * 8 u32 = 64 uint4
                bufs[t].q[1] = bbase[t * 64 + 1];
            }
        }

        // ---- stage A tile: each wave stages exactly its own 16-row band ----
        {
            int row  = tid >> 1;          // 0..127 (wave w covers rows 16w..16w+15)
            int half = tid & 1;           // which 16-float segment of the 32-K slice
            int grow = mBase + row;
            int srow = grow < NN ? grow : NN - 1;
            const float4* src = (const float4*)(X + (size_t)srow * KD + ks * 32 + half * 16);
            float4 f0 = src[0], f1 = src[1], f2 = src[2], f3 = src[3];
            if (grow >= NN) {
                f0 = make_float4(0.f, 0.f, 0.f, 0.f);
                f1 = f0; f2 = f0; f3 = f0;
            }
            unsigned* dst = &aLds[row * 16 + half * 8];
            dst[0] = pack_bf2(f0.x, f0.y); dst[1] = pack_bf2(f0.z, f0.w);
            dst[2] = pack_bf2(f1.x, f1.y); dst[3] = pack_bf2(f1.z, f1.w);
            dst[4] = pack_bf2(f2.x, f2.y); dst[5] = pack_bf2(f2.z, f2.w);
            dst[6] = pack_bf2(f3.x, f3.y); dst[7] = pack_bf2(f3.z, f3.w);
        }

        // ---- A fragment: two contiguous 16B runs, same contiguous k order as B (k = 16h + s) ----
        union { v16bf v; uint4 q[2]; } af;
        {
            const int abase = arow * 16 + pbase;
            af.q[0] = *(const uint4*)&aLds[abase];       // pairs pbase+0..3
            af.q[1] = *(const uint4*)&aLds[abase + 4];   // pairs pbase+4..7
        }

        // ---- 11 WMMAs back-to-back ----
        #pragma unroll
        for (int t = 0; t < NT_TILES; ++t)
            acc[t] = __builtin_amdgcn_wmma_f32_16x16x32_bf16(
                false, af.v, false, bufs[t].v, (short)0, acc[t], false, false);
    }

    // ---- store C: lanes 0-15 rows r, lanes 16-31 rows 8+r ----
    int crowBase = mBase + (wave << 4) + ((lane >= 16) ? 8 : 0);
    int ccol = lane & 15;
    #pragma unroll
    for (int t = 0; t < NT_TILES; ++t)
        #pragma unroll
        for (int r = 0; r < 8; ++r) {
            int row = crowBase + r;
            if (row < NN) C[(size_t)row * NCAT + t * 16 + ccol] = acc[t][r];
        }
}

// ---------------- K3: per-edge score max (softmax stabilization) ----------------
__global__ __launch_bounds__(256)
void edge_max_kernel(const int* __restrict__ src, const int* __restrict__ dst,
                     const float* __restrict__ Hs, const float* __restrict__ Hd,
                     float* __restrict__ emax) {
    int idx = blockIdx.x * blockDim.x + threadIdx.x;
    if (idx >= EE * HH) return;
    int e = idx >> 2, h = idx & 3;
    int s = src[e], d = dst[e];
    float v = Hs[(size_t)s * NCAT + 160 + h] + Hd[(size_t)d * NCAT + 164 + h];
    v = v > 0.f ? v : 0.2f * v;             // LeakyReLU(0.2)
    atomicMaxFloat(&emax[(size_t)d * HH + h], v);
}

// ---------------- K4: per-edge softmax weight + message scatter (1 wave32/edge) ----------------
__global__ __launch_bounds__(256)
void edge_msg_kernel(const int* __restrict__ src, const int* __restrict__ dst,
                     const float* __restrict__ Hs, const float* __restrict__ Hd,
                     const float* __restrict__ emax, float* __restrict__ denom,
                     float* __restrict__ agg) {
    int gw   = (blockIdx.x * blockDim.x + threadIdx.x) >> 5;   // edge id (wave32)
    int lane = threadIdx.x & 31;
    if (gw >= EE) return;
    int s = src[gw], d = dst[gw];
    // prefetch the next edge-group's source row (gfx1250 global_prefetch path)
    if (gw + 8 < EE) {
        int s2 = src[gw + 8];
        __builtin_prefetch(Hs + (size_t)s2 * NCAT + lane * 4, 0, 0);
    }
    const float* hsrow = Hs + (size_t)s * NCAT;
    const float* elp   = hsrow + 160;
    const float* erp   = Hd + (size_t)d * NCAT + 164;
    const float* emx   = emax + (size_t)d * HH;
    float ex[HH];
    #pragma unroll
    for (int h = 0; h < HH; ++h) {
        float e = elp[h] + erp[h];
        e = e > 0.f ? e : 0.2f * e;
        ex[h] = __expf(e - emx[h]);
    }
    if (lane < HH) atomicAdd(&denom[(size_t)d * HH + lane], ex[lane]);
    float* arow = agg + (size_t)d * HD;
    #pragma unroll
    for (int h = 0; h < HH; ++h)
        atomicAdd(&arow[h * DO + lane], ex[h] * hsrow[h * DO + lane]);
}

// ---------------- K5: finalize (divide, elu, + self-loop, elu) in place ----------------
__global__ __launch_bounds__(256)
void finalize_kernel(float* __restrict__ out,
                     const float* __restrict__ den_item, const float* __restrict__ den_user,
                     const float* __restrict__ Uw, const float* __restrict__ Iw) {
    int idx = blockIdx.x * blockDim.x + threadIdx.x;
    const int half = NN * HD;
    if (idx >= 2 * half) return;
    bool isItem = idx >= half;
    int r  = isItem ? idx - half : idx;
    int n  = r >> 7;
    int c  = r & 127;
    int dd = c & 31;
    int h  = c >> 5;
    float den = isItem ? den_item[(size_t)n * HH + h] : den_user[(size_t)n * HH + h];
    const float* L = isItem ? Iw : Uw;
    float loopv = L[(size_t)n * NCAT + 128 + dd];
    float a = out[idx] / fmaxf(den, 1e-9f);
    a = eluf(a);          // GATConv activation
    a = eluf(a + loopv);  // self-loop fusion + wrapper activation
    out[idx] = a;
}

// ---------------- launch ----------------
extern "C" void kernel_launch(void* const* d_in, const int* in_sizes, int n_in,
                              void* d_out, int out_size, void* d_ws, size_t ws_size,
                              hipStream_t stream) {
    const float* feat_user = (const float*)d_in[0];
    const float* feat_item = (const float*)d_in[1];
    const int*   src_r1    = (const int*)d_in[2];
    const int*   dst_r1    = (const int*)d_in[3];
    const int*   src_r2    = (const int*)d_in[4];
    const int*   dst_r2    = (const int*)d_in[5];
    const float* fc_r1     = (const float*)d_in[6];
    const float* attn_l_r1 = (const float*)d_in[7];
    const float* attn_r_r1 = (const float*)d_in[8];
    const float* fc_r2     = (const float*)d_in[9];
    const float* attn_l_r2 = (const float*)d_in[10];
    const float* attn_r_r2 = (const float*)d_in[11];
    const float* loop_w    = (const float*)d_in[12];

    float* out    = (float*)d_out;
    float* h_user = out;                       // output slot 0
    float* h_item = out + (size_t)NN * HD;     // output slot 1

    // workspace carve-up (floats)
    float*    U     = (float*)d_ws;                                // NN x NCAT
    float*    I     = U + (size_t)NN * NCAT;                       // NN x NCAT
    unsigned* Wp    = (unsigned*)(I + (size_t)NN * NCAT);          // 2 x PNT
    float*    emax1 = (float*)(Wp + 2 * PNT);                      // NN x H (item dst)
    float*    emax2 = emax1 + (size_t)NN * HH;                     // NN x H (user dst)
    float*    den1  = emax2 + (size_t)NN * HH;
    float*    den2  = den1  + (size_t)NN * HH;

    // 0) pack cat-weight matrices into WMMA B-fragment order (bf16 pairs)
    {
        int total = 2 * PNT;
        pack_w_kernel<<<(total + 255) / 256, 256, 0, stream>>>(
            fc_r1, fc_r2, attn_l_r1, attn_r_r1, attn_l_r2, attn_r_r2, loop_w, Wp);
    }
    // 1) zero output accumulators, init softmax stats
    {
        int total = 2 * NN * HD;
        init_kernel<<<(total + 255) / 256, 256, 0, stream>>>(out, emax1, emax2, den1, den2);
    }
    // 2) WMMA GEMMs: U = feat_user @ Wcat_u, I = feat_item @ Wcat_i
    {
        dim3 grid((NN + 127) / 128, 2);
        gemm_cat_kernel<<<grid, 256, 0, stream>>>(feat_user, feat_item, Wp, U, I);
    }
    // 3) edge softmax max per relation
    {
        int total = EE * HH;
        edge_max_kernel<<<(total + 255) / 256, 256, 0, stream>>>(src_r1, dst_r1, U, I, emax1);
        edge_max_kernel<<<(total + 255) / 256, 256, 0, stream>>>(src_r2, dst_r2, I, U, emax2);
    }
    // 4) edge messages: one wave32 per edge, atomic scatter into d_out halves
    {
        int blocks = (EE * 32 + 255) / 256;
        edge_msg_kernel<<<blocks, 256, 0, stream>>>(src_r1, dst_r1, U, I, emax1, den1, h_item);
        edge_msg_kernel<<<blocks, 256, 0, stream>>>(src_r2, dst_r2, I, U, emax2, den2, h_user);
    }
    // 5) finalize in place
    {
        int total = 2 * NN * HD;
        finalize_kernel<<<(total + 255) / 256, 256, 0, stream>>>(out, den1, den2, U, I);
    }
}
